// DenseGNN_8014408974713
// MI455X (gfx1250) — matrix operand order repaired and checked by reading, not compiled, not run on hardware
//
#include <hip/hip_runtime.h>

#define NN 10000
#define EE 100000
#define GG 64
#define AFEAT 92
#define RBF_N 80
#define EMB 64
#define HH 256
#define PDIM 64
#define NLAYER 4

typedef __attribute__((ext_vector_type(16))) __bf16 v16bf;
typedef __attribute__((ext_vector_type(8)))  float v8f;

union BFrag { v16bf v; unsigned short u[16]; };
union O8 { unsigned short h[8]; uint4 q; };

__device__ __forceinline__ unsigned short f2bf(float f) {
  unsigned int x = __float_as_uint(f);
  x += 0x7fffu + ((x >> 16) & 1u);   // round to nearest even
  return (unsigned short)(x >> 16);
}

// ---------------------------------------------------------------------------
// Pack f32 weight W[K,N] (row-major) into WMMA B-fragment-major bf16 layout.
// Tile (kt,nt) covers K rows [kt*32,kt*32+32), cols [nt*16,nt*16+16).
// Lane l (0..31): col = nt*16 + (l&15); h = l>>4; k = kt*32 + 8h + i (i<8), kt*32 + 16 + 8h + (i-8) (i>=8).
// Each lane's 16 halves are stored contiguously -> fragment load is 2x b128.
// ---------------------------------------------------------------------------
__global__ void pack_weights(const float* __restrict__ W, unsigned short* __restrict__ out,
                             int K, int N) {
  int tile = blockIdx.x;
  int ntiles = N >> 4;
  int kt = tile / ntiles, nt = tile - kt * ntiles;
  int lane = threadIdx.x;
  int col = nt * 16 + (lane & 15);
  int ks  = kt * 32 + ((lane >> 4) << 3);
  unsigned short* o = out + (size_t)tile * 512 + (size_t)lane * 16;
#pragma unroll
  for (int i = 0; i < 16; ++i) {
    int k = ks + (i < 8 ? i : 8 + i);
    float v = (k < K) ? W[(size_t)k * N + col] : 0.f;
    o[i] = f2bf(v);
  }
}

// ---------------------------------------------------------------------------
// Optimized GEMM for N=256: C = silu(bn(A @ W + bias) [+ resid])
// Block: 256 threads (8 waves), tile 32 rows x 256 cols.
// Each wave: 32 rows x 32 cols -> 2 A-frags x 2 B-frags -> 4 WMMA / K-chunk,
// software-pipelined (fragments for kt+1 load while WMMAs for kt execute).
// MODE 0: A = A0[M,K] row-major (zero padded to Kpad)
// MODE 1: edge gather: row e = concat[ n[src[e]], n[dst[e]], efeat[e] ], K=768
// MODE 2: node gather: row v = concat[ n[v], agg[v]/max(deg[v],1) ],    K=512
// ---------------------------------------------------------------------------
template <int MODE>
__global__ __launch_bounds__(256)
void wmma_gemm256(const float* __restrict__ A0, const float* __restrict__ A1,
                  const int* __restrict__ srcI, const int* __restrict__ dstI,
                  const float* __restrict__ deg,
                  const unsigned short* __restrict__ Bp,
                  const float* __restrict__ bias, const float* __restrict__ gamma,
                  const float* __restrict__ beta, const float* __restrict__ resid,
                  float* __restrict__ C, int M, int K, int Kpad) {
  __shared__ __align__(16) unsigned short Asb[32 * 768];   // up to 48 KB bf16 A tile
  const int row0 = blockIdx.x * 32;

  // ---- stage A tile (32 x Kpad) into LDS as bf16, 8 elements per thread-step:
  //      2x global_load_b128 -> convert -> 1x ds_store_b128, gather fused. ----
  const int kc8 = Kpad >> 3;
  for (int idx = threadIdx.x; idx < 32 * kc8; idx += 256) {
    int r = idx / kc8, k = (idx - r * kc8) << 3;
    int m = row0 + r;
    const float* sp = nullptr;
    float scale = 1.f;
    if (m < M) {
      if (MODE == 0) {
        if (k + 8 <= K) sp = A0 + (size_t)m * K + k;        // else: tail path below
      } else if (MODE == 1) {                               // segments are 256-aligned
        if (k < HH)          sp = A0 + (size_t)srcI[m] * HH + k;
        else if (k < 2 * HH) sp = A0 + (size_t)dstI[m] * HH + (k - HH);
        else                 sp = A1 + (size_t)m * HH + (k - 2 * HH);
      } else {
        if (k < HH)  sp = A0 + (size_t)m * HH + k;
        else { sp = A1 + (size_t)m * HH + (k - HH); scale = 1.f / fmaxf(deg[m], 1.f); }
      }
    }
    O8 o;
    if (sp) {
      const float4* p4 = reinterpret_cast<const float4*>(sp);
      float4 f0 = p4[0], f1 = p4[1];
      o.h[0] = f2bf(f0.x * scale); o.h[1] = f2bf(f0.y * scale);
      o.h[2] = f2bf(f0.z * scale); o.h[3] = f2bf(f0.w * scale);
      o.h[4] = f2bf(f1.x * scale); o.h[5] = f2bf(f1.y * scale);
      o.h[6] = f2bf(f1.z * scale); o.h[7] = f2bf(f1.w * scale);
    } else {
#pragma unroll
      for (int i = 0; i < 8; ++i) {
        float v = 0.f;
        if (MODE == 0 && m < M && (k + i) < K) v = A0[(size_t)m * K + k + i];
        o.h[i] = f2bf(v);
      }
    }
    *reinterpret_cast<uint4*>(&Asb[(size_t)r * Kpad + k]) = o.q;
  }
  __syncthreads();

  const int wave = threadIdx.x >> 5;
  const int lane = threadIdx.x & 31;
  const int hs   = lane >> 4;      // half-wave select
  const int lr   = lane & 15;
  const int nchunk = Kpad >> 5;

  v8f acc00 = {0.f,0.f,0.f,0.f,0.f,0.f,0.f,0.f};
  v8f acc01 = acc00, acc10 = acc00, acc11 = acc00;

  const unsigned short* arow = &Asb[lr * Kpad + hs * 8];        // rows 0..15
  const unsigned short* bcol = Bp + (size_t)(wave * 2) * 512 + (size_t)lane * 16;

  auto loadFrags = [&](int kt, BFrag& fa0, BFrag& fa1, BFrag& fb0, BFrag& fb1) {
    const unsigned short* ap0 = arow + kt * 32;
    const unsigned short* ap1 = ap0 + 16 * Kpad;                // rows 16..31
#pragma unroll
    for (int i = 0; i < 8; ++i) {
      fa0.u[i] = ap0[i]; fa0.u[8 + i] = ap0[16 + i];
      fa1.u[i] = ap1[i]; fa1.u[8 + i] = ap1[16 + i];
    }
    const unsigned short* bp = bcol + (size_t)kt * 16 * 512;
#pragma unroll
    for (int i = 0; i < 16; ++i) { fb0.u[i] = bp[i]; fb1.u[i] = bp[512 + i]; }
  };

  BFrag a0, a1, b0, b1;
  loadFrags(0, a0, a1, b0, b1);
  for (int kt = 0; kt < nchunk; ++kt) {
    BFrag c0 = a0, c1 = a1, d0 = b0, d1 = b1;   // current (register rotation)
    if (kt + 1 < nchunk) loadFrags(kt + 1, a0, a1, b0, b1);   // prefetch next
    acc00 = __builtin_amdgcn_wmma_f32_16x16x32_bf16(false, c0.v, false, d0.v, (short)0, acc00, false, false);
    acc01 = __builtin_amdgcn_wmma_f32_16x16x32_bf16(false, c0.v, false, d1.v, (short)0, acc01, false, false);
    acc10 = __builtin_amdgcn_wmma_f32_16x16x32_bf16(false, c1.v, false, d0.v, (short)0, acc10, false, false);
    acc11 = __builtin_amdgcn_wmma_f32_16x16x32_bf16(false, c1.v, false, d1.v, (short)0, acc11, false, false);
  }

  // ---- fused epilogue: bias + eval-BN (+ residual) + SiLU ----
  const float rs = rsqrtf(1.0f + 1e-5f);
#pragma unroll
  for (int cj = 0; cj < 2; ++cj) {
    int col = (wave * 2 + cj) * 16 + lr;
    float s = gamma[col] * rs;
    float t = bias[col] * s + beta[col];
#pragma unroll
    for (int rj = 0; rj < 2; ++rj) {
      v8f acc = (rj == 0) ? ((cj == 0) ? acc00 : acc01)
                          : ((cj == 0) ? acc10 : acc11);
      int mb = row0 + rj * 16 + hs * 8;
#pragma unroll
      for (int jj = 0; jj < 8; ++jj) {
        int m = mb + jj;
        if (m < M) {
          float y = acc[jj] * s + t;
          if (resid) y += resid[(size_t)m * HH + col];
          y = y * (1.f / (1.f + __expf(-y)));   // SiLU
          C[(size_t)m * HH + col] = y;
        }
      }
    }
  }
}

// ---------------------------------------------------------------------------
// Generic small GEMM (used only for the tiny e1 layer, N=64):
// C[M,N] = silu(bn(A0 @ W + bias)), A0[M,K] row-major.
// ---------------------------------------------------------------------------
__global__ __launch_bounds__(256)
void wmma_gemm_small(const float* __restrict__ A0,
                     const unsigned short* __restrict__ Bp,
                     const float* __restrict__ bias, const float* __restrict__ gamma,
                     const float* __restrict__ beta,
                     float* __restrict__ C, int M, int N, int K, int Kpad) {
  __shared__ __align__(16) unsigned short Asb[16 * 768];
  const int row0 = blockIdx.x * 16;
  for (int idx = threadIdx.x; idx < 16 * Kpad; idx += 256) {
    int r = idx / Kpad, k = idx - r * Kpad;
    int m = row0 + r;
    float v = 0.f;
    if (m < M && k < K) v = A0[(size_t)m * K + k];
    Asb[r * Kpad + k] = f2bf(v);
  }
  __syncthreads();

  const int wave = threadIdx.x >> 5;
  const int lane = threadIdx.x & 31;
  const int hs = lane >> 4, lr = lane & 15;
  const int nchunk = Kpad >> 5;
  const int ntiles = N >> 4;

  for (int nt = wave; nt < ntiles; nt += 8) {
    v8f acc = {0.f,0.f,0.f,0.f,0.f,0.f,0.f,0.f};
    const unsigned short* bbase = Bp + (size_t)nt * 512 + (size_t)lane * 16;
    for (int kt = 0; kt < nchunk; ++kt) {
      BFrag a, b;
      const unsigned short* ap = &Asb[lr * Kpad + kt * 32 + hs * 8];
#pragma unroll
      for (int i = 0; i < 8; ++i) { a.u[i] = ap[i]; a.u[8 + i] = ap[16 + i]; }
      const unsigned short* bp = bbase + (size_t)kt * ntiles * 512;
#pragma unroll
      for (int i = 0; i < 16; ++i) b.u[i] = bp[i];
      acc = __builtin_amdgcn_wmma_f32_16x16x32_bf16(false, a.v, false, b.v,
                                                    (short)0, acc, false, false);
    }
    int col = nt * 16 + lr;
    float s = gamma[col] * rsqrtf(1.0f + 1e-5f);
    float t = bias[col] * s + beta[col];
#pragma unroll
    for (int j = 0; j < 8; ++j) {
      int m = row0 + j + hs * 8;
      if (m < M) {
        float y = acc[j] * s + t;
        y = y * (1.f / (1.f + __expf(-y)));
        C[(size_t)m * N + col] = y;
      }
    }
  }
}

// ---------------------------------------------------------------------------
__global__ void zero_f32(float* p, long n) {
  long i = (long)blockIdx.x * blockDim.x + threadIdx.x;
  long stride = (long)gridDim.x * blockDim.x;
  for (; i < n; i += stride) p[i] = 0.f;
}

__global__ void rbf_kernel(const float* __restrict__ bondlen, float* __restrict__ rbf) {
  long idx = (long)blockIdx.x * blockDim.x + threadIdx.x;
  if (idx >= (long)EE * RBF_N) return;
  int e = (int)(idx / RBF_N), j = (int)(idx - (long)e * RBF_N);
  const float ls = 8.0f / (RBF_N - 1);
  float d = (bondlen[e] - j * ls) / ls;
  rbf[idx] = __expf(-d * d);
}

__global__ void deg_kernel(const int* __restrict__ dst, float* deg) {
  int e = blockIdx.x * blockDim.x + threadIdx.x;
  if (e < EE) atomicAdd(&deg[dst[e]], 1.0f);
}

__global__ void agg_kernel(const float* __restrict__ ue, const int* __restrict__ dst,
                           float* __restrict__ agg) {
  long idx = (long)blockIdx.x * blockDim.x + threadIdx.x;
  if (idx >= (long)EE * HH) return;
  int e = (int)(idx >> 8), c = (int)(idx & (HH - 1));
  atomicAdd(&agg[(size_t)dst[e] * HH + c], ue[idx]);
}

__global__ void gcnt_kernel(const int* __restrict__ gid, float* gcnt) {
  int v = blockIdx.x * blockDim.x + threadIdx.x;
  if (v < NN) atomicAdd(&gcnt[gid[v]], 1.0f);
}

__global__ void pool_kernel(const float* __restrict__ n, const int* __restrict__ gid,
                            float* __restrict__ gsum) {
  long idx = (long)blockIdx.x * blockDim.x + threadIdx.x;
  if (idx >= (long)NN * HH) return;
  int v = (int)(idx >> 8), c = (int)(idx & (HH - 1));
  atomicAdd(&gsum[(size_t)gid[v] * HH + c], n[idx]);
}

// ProjectionHead + LayerNorm. One block per graph, PDIM threads.
__global__ void head_kernel(const float* __restrict__ gsum, const float* __restrict__ gcnt,
                            const float* __restrict__ pw, const float* __restrict__ pb,
                            const float* __restrict__ fw, const float* __restrict__ fb,
                            const float* __restrict__ lng, const float* __restrict__ lnb,
                            float* __restrict__ out) {
  int g = blockIdx.x, t = threadIdx.x;
  __shared__ float gf[HH];
  __shared__ float proj[PDIM];
  __shared__ float hbuf[PDIM];
  __shared__ float red[PDIM];
  float inv = 1.f / fmaxf(gcnt[g], 1.f);
  for (int i = t; i < HH; i += PDIM) gf[i] = gsum[(size_t)g * HH + i] * inv;
  __syncthreads();
  float acc = pb[t];
  for (int h = 0; h < HH; ++h) acc += gf[h] * pw[h * PDIM + t];
  proj[t] = acc;
  float x3 = acc * acc * acc;
  hbuf[t] = 0.5f * acc * (1.f + tanhf(0.7978845608f * (acc + 0.044715f * x3)));
  __syncthreads();
  float a2 = fb[t];
  for (int h = 0; h < PDIM; ++h) a2 += hbuf[h] * fw[h * PDIM + t];
  a2 += proj[t];
  red[t] = a2;
  __syncthreads();
  float mu = 0.f;
  for (int i = 0; i < PDIM; ++i) mu += red[i];
  mu *= (1.f / PDIM);
  float var = 0.f;
  for (int i = 0; i < PDIM; ++i) { float d = red[i] - mu; var += d * d; }
  var *= (1.f / PDIM);
  out[(size_t)g * PDIM + t] = (a2 - mu) * rsqrtf(var + 1e-5f) * lng[t] + lnb[t];
}

// ---------------------------------------------------------------------------
extern "C" void kernel_launch(void* const* d_in, const int* in_sizes, int n_in,
                              void* d_out, int out_size, void* d_ws, size_t ws_size,
                              hipStream_t stream) {
  (void)in_sizes; (void)n_in; (void)out_size; (void)ws_size;
  // ---- inputs (setup_inputs dict order, params flattened in insertion order) ----
  const float* atom_feats = (const float*)d_in[0];
  const float* bondlen    = (const float*)d_in[1];
  const int*   src        = (const int*)d_in[2];
  const int*   dst        = (const int*)d_in[3];
  const int*   gid        = (const int*)d_in[4];
  const float* atom_w = (const float*)d_in[5];
  const float* atom_b = (const float*)d_in[6];
  const float* atom_g = (const float*)d_in[7];
  const float* atom_be= (const float*)d_in[8];
  const float* e1_w = (const float*)d_in[9];
  const float* e1_b = (const float*)d_in[10];
  const float* e1_g = (const float*)d_in[11];
  const float* e1_be= (const float*)d_in[12];
  const float* e2_w = (const float*)d_in[13];
  const float* e2_b = (const float*)d_in[14];
  const float* e2_g = (const float*)d_in[15];
  const float* e2_be= (const float*)d_in[16];
  const float* ew1 = (const float*)d_in[17];
  const float* ew2 = (const float*)d_in[18];
  const float* nw1 = (const float*)d_in[19];
  const float* nw2 = (const float*)d_in[20];
  const float* eb1 = (const float*)d_in[21];
  const float* eb2 = (const float*)d_in[22];
  const float* nb1 = (const float*)d_in[23];
  const float* nb2 = (const float*)d_in[24];
  const float* ebe1= (const float*)d_in[25];
  const float* ebe2= (const float*)d_in[26];
  const float* nbe1= (const float*)d_in[27];
  const float* nbe2= (const float*)d_in[28];
  const float* eg1 = (const float*)d_in[29];
  const float* eg2 = (const float*)d_in[30];
  const float* ng1 = (const float*)d_in[31];
  const float* ng2 = (const float*)d_in[32];
  const float* pw  = (const float*)d_in[33];
  const float* pb  = (const float*)d_in[34];
  const float* fw  = (const float*)d_in[35];
  const float* fb  = (const float*)d_in[36];
  const float* lng = (const float*)d_in[37];
  const float* lnb = (const float*)d_in[38];

  // ---- workspace carve-up ----
  char* ws = (char*)d_ws;
  size_t off = 0;
  auto alloc = [&](size_t bytes) -> char* {
    off = (off + 255) & ~(size_t)255;
    char* p = ws + off;
    off += bytes;
    return p;
  };
  unsigned short* pk_atom = (unsigned short*)alloc((size_t)3 * 16 * 512 * 2);   // K92->96
  unsigned short* pk_e1   = (unsigned short*)alloc((size_t)3 * 4  * 512 * 2);   // K80->96,N64
  unsigned short* pk_e2   = (unsigned short*)alloc((size_t)2 * 16 * 512 * 2);   // K64
  unsigned short* pk_ew1  = (unsigned short*)alloc((size_t)NLAYER * 24 * 16 * 512 * 2); // K768
  unsigned short* pk_ew2  = (unsigned short*)alloc((size_t)NLAYER * 8  * 16 * 512 * 2); // K256
  unsigned short* pk_nw1  = (unsigned short*)alloc((size_t)NLAYER * 16 * 16 * 512 * 2); // K512 (pad block dropped)
  unsigned short* pk_nw2  = (unsigned short*)alloc((size_t)NLAYER * 8  * 16 * 512 * 2); // K256
  float* n_a  = (float*)alloc((size_t)NN * HH * 4);
  float* n_b  = (float*)alloc((size_t)NN * HH * 4);
  float* e_a  = (float*)alloc((size_t)EE * HH * 4);
  float* e_b  = (float*)alloc((size_t)EE * HH * 4);   // also holds RBF [E,80] initially
  float* mid  = (float*)alloc((size_t)EE * HH * 4);   // MLP hidden (edge- and node-sized)
  float* agg  = (float*)alloc((size_t)NN * HH * 4);
  float* deg  = (float*)alloc((size_t)NN * 4);
  float* gsum = (float*)alloc((size_t)GG * HH * 4);
  float* gcnt = (float*)alloc((size_t)GG * 4);

  // ---- pack all weights to WMMA fragment layout ----
  pack_weights<<<3 * 16, 32, 0, stream>>>(atom_w, pk_atom, AFEAT, HH);
  pack_weights<<<3 * 4, 32, 0, stream>>>(e1_w, pk_e1, RBF_N, EMB);
  pack_weights<<<2 * 16, 32, 0, stream>>>(e2_w, pk_e2, EMB, HH);
  for (int l = 0; l < NLAYER; ++l) {
    pack_weights<<<24 * 16, 32, 0, stream>>>(ew1 + (size_t)l * 3 * HH * HH,
                                             pk_ew1 + (size_t)l * 24 * 16 * 512, 3 * HH, HH);
    pack_weights<<<8 * 16, 32, 0, stream>>>(ew2 + (size_t)l * HH * HH,
                                            pk_ew2 + (size_t)l * 8 * 16 * 512, HH, HH);
    pack_weights<<<16 * 16, 32, 0, stream>>>(nw1 + (size_t)l * 3 * HH * HH,
                                             pk_nw1 + (size_t)l * 16 * 16 * 512, 2 * HH, HH);
    pack_weights<<<8 * 16, 32, 0, stream>>>(nw2 + (size_t)l * HH * HH,
                                            pk_nw2 + (size_t)l * 8 * 16 * 512, HH, HH);
  }

  // ---- degree ----
  zero_f32<<<64, 256, 0, stream>>>(deg, NN);
  deg_kernel<<<(EE + 255) / 256, 256, 0, stream>>>(dst, deg);

  // ---- embeddings ----
  rbf_kernel<<<(int)(((long)EE * RBF_N + 255) / 256), 256, 0, stream>>>(bondlen, e_b);
  // e1: [E,80] -> [E,64]
  wmma_gemm_small<<<EE / 16, 256, 0, stream>>>(e_b, pk_e1, e1_b, e1_g, e1_be,
                                               mid, EE, EMB, RBF_N, 96);
  // e2: [E,64] -> [E,256]
  wmma_gemm256<0><<<(EE + 31) / 32, 256, 0, stream>>>(mid, nullptr, nullptr, nullptr, nullptr,
      pk_e2, e2_b, e2_g, e2_be, nullptr, e_a, EE, EMB, 64);
  // atoms: [N,92] -> [N,256]
  wmma_gemm256<0><<<(NN + 31) / 32, 256, 0, stream>>>(atom_feats, nullptr, nullptr, nullptr, nullptr,
      pk_atom, atom_b, atom_g, atom_be, nullptr, n_a, NN, AFEAT, 96);

  // ---- DenseGNN layers ----
  float *nc = n_a, *nx = n_b, *ec = e_a, *ex = e_b;
  for (int l = 0; l < NLAYER; ++l) {
    const float* Eb1 = eb1 + l * HH; const float* Eg1 = eg1 + l * HH; const float* Ebe1 = ebe1 + l * HH;
    const float* Eb2 = eb2 + l * HH; const float* Eg2 = eg2 + l * HH; const float* Ebe2 = ebe2 + l * HH;
    const float* Nb1 = nb1 + l * HH; const float* Ng1 = ng1 + l * HH; const float* Nbe1 = nbe1 + l * HH;
    const float* Nb2 = nb2 + l * HH; const float* Ng2 = ng2 + l * HH; const float* Nbe2 = nbe2 + l * HH;
    // edge MLP layer 1: gathered concat[n[src],n[dst],e] @ ew1, K=768
    wmma_gemm256<1><<<(EE + 31) / 32, 256, 0, stream>>>(nc, ec, src, dst, nullptr,
        pk_ew1 + (size_t)l * 24 * 16 * 512, Eb1, Eg1, Ebe1, nullptr, mid, EE, 3 * HH, 3 * HH);
    // edge MLP layer 2 + residual e + SiLU -> new edge feats
    wmma_gemm256<0><<<(EE + 31) / 32, 256, 0, stream>>>(mid, nullptr, nullptr, nullptr, nullptr,
        pk_ew2 + (size_t)l * 8 * 16 * 512, Eb2, Eg2, Ebe2, ec, ex, EE, HH, HH);
    // mean aggregation at dst
    zero_f32<<<512, 256, 0, stream>>>(agg, (long)NN * HH);
    agg_kernel<<<(int)(((long)EE * HH + 255) / 256), 256, 0, stream>>>(ex, dst, agg);
    // node MLP layer 1: gathered concat[n, agg/deg] @ nw1[:512], K=512 (pad block = 0)
    wmma_gemm256<2><<<(NN + 31) / 32, 256, 0, stream>>>(nc, agg, nullptr, nullptr, deg,
        pk_nw1 + (size_t)l * 16 * 16 * 512, Nb1, Ng1, Nbe1, nullptr, mid, NN, 2 * HH, 2 * HH);
    // node MLP layer 2 + residual n + SiLU -> new node feats
    wmma_gemm256<0><<<(NN + 31) / 32, 256, 0, stream>>>(mid, nullptr, nullptr, nullptr, nullptr,
        pk_nw2 + (size_t)l * 8 * 16 * 512, Nb2, Ng2, Nbe2, nc, nx, NN, HH, HH);
    float* t;
    t = nc; nc = nx; nx = t;
    t = ec; ec = ex; ex = t;
  }

  // ---- readout + projection head ----
  zero_f32<<<64, 256, 0, stream>>>(gsum, (long)GG * HH);
  zero_f32<<<1, 64, 0, stream>>>(gcnt, GG);
  gcnt_kernel<<<(NN + 255) / 256, 256, 0, stream>>>(gid, gcnt);
  pool_kernel<<<(int)(((long)NN * HH + 255) / 256), 256, 0, stream>>>(nc, gid, gsum);
  head_kernel<<<GG, PDIM, 0, stream>>>(gsum, gcnt, pw, pb, fw, fb, lng, lnb, (float*)d_out);
}
